// SparseSelfAttention_51969104281854
// MI455X (gfx1250) — matrix operand order repaired and checked by reading, not compiled, not run on hardware
//
#include <hip/hip_runtime.h>

// ---------------------------------------------------------------------------
// SparseSelfAttention on MI455X (gfx1250): bf16 WMMA everywhere, f32 accum.
// B=2, S=2048, D=1024, H=16, HD=64, top-k K = int(2048*0.2) = 409.
//
// Pipeline:
//   1) f32 -> bf16 conversion of X and the four weight matrices
//   2) fused QKV projection GEMM, 64x64 register tile per wave
//      (Q,K stored head-major [B,H,S,HD]; V stored transposed [B,H,HD,S])
//   3) attention: scores->LDS, bisection top-k + masked softmax, P@V
//   4) output projection GEMM -> f32
// ---------------------------------------------------------------------------

#define BB   2
#define SS   2048
#define DD   1024
#define HH   16
#define HDD  64
#define KK   409            // max(int(S*(1-0.8)),1)

typedef __attribute__((ext_vector_type(16))) __bf16 v16bf;
typedef __attribute__((ext_vector_type(8)))  __bf16 v8bf;
typedef __attribute__((ext_vector_type(8)))  float  v8f;

static __device__ __forceinline__ v16bf cat16(v8bf lo, v8bf hi) {
    return __builtin_shufflevector(lo, hi, 0,1,2,3,4,5,6,7,8,9,10,11,12,13,14,15);
}

static __device__ __forceinline__ v8f wmma_bf16(v16bf a, v16bf b, v8f c) {
    // v_wmma_f32_16x16x32_bf16
    return __builtin_amdgcn_wmma_f32_16x16x32_bf16(
        /*neg_a=*/false, a, /*neg_b=*/false, b,
        /*c_mod=*/(short)0, c, /*reuse_a=*/false, /*reuse_b=*/false);
}

// A-fragment (16x32 bf16, M rows = base..base+15):
// lane<16: row base+l, k {0..7,16..23}; lane>=16: same rows, k {8..15,24..31}
static __device__ __forceinline__ v16bf load_afrag(const __bf16* rowptr) {
    return cat16(*(const v8bf*)(rowptr), *(const v8bf*)(rowptr + 16));
}
// B-fragment (32x16): lane = column n, lane<16 k {0..7,16..23}, lane>=16 k {8..15,24..31},
// along the source row.
static __device__ __forceinline__ v16bf load_bfrag(const __bf16* rowptr) {
    return cat16(*(const v8bf*)(rowptr), *(const v8bf*)(rowptr + 16));
}

// wave32 butterfly reductions
static __device__ __forceinline__ float wred_max(float v) {
    #pragma unroll
    for (int m = 16; m >= 1; m >>= 1) v = fmaxf(v, __shfl_xor(v, m, 32));
    return v;
}
static __device__ __forceinline__ float wred_min(float v) {
    #pragma unroll
    for (int m = 16; m >= 1; m >>= 1) v = fminf(v, __shfl_xor(v, m, 32));
    return v;
}
static __device__ __forceinline__ float wred_sum(float v) {
    #pragma unroll
    for (int m = 16; m >= 1; m >>= 1) v += __shfl_xor(v, m, 32);
    return v;
}
static __device__ __forceinline__ int wred_sum_i(int v) {
    #pragma unroll
    for (int m = 16; m >= 1; m >>= 1) v += __shfl_xor(v, m, 32);
    return v;
}

// ---------------------------------------------------------------------------
// f32 -> bf16 conversion
// ---------------------------------------------------------------------------
__global__ void cvt_f32_bf16(const float* __restrict__ src,
                             __bf16* __restrict__ dst, int n) {
    int i = blockIdx.x * blockDim.x + threadIdx.x;
    if (i < n) dst[i] = (__bf16)src[i];
}

// ---------------------------------------------------------------------------
// Fused QKV projection: C[m,n] = sum_k X[m,k]*W[n,k] + bias[n]
// M=4096, N=1024, K=1024.  64x64 register tile per wave:
// 4 A-frags x 4 B-frags -> 16 WMMA per 32-wide K-step (32 flops/byte).
// blockIdx.z selects (Wq,Q) / (Wk,K) / (Wv,V-transposed).
// ---------------------------------------------------------------------------
__global__ __launch_bounds__(256) void
gemm_qkv_fused(const __bf16* __restrict__ X,
               const __bf16* __restrict__ Wq, const __bf16* __restrict__ Wk,
               const __bf16* __restrict__ Wv,
               const float* __restrict__ bq, const float* __restrict__ bk,
               const float* __restrict__ bv,
               __bf16* __restrict__ Qo, __bf16* __restrict__ Ko,
               __bf16* __restrict__ Vo) {
    const int z = blockIdx.z;
    const __bf16* W    = (z == 0) ? Wq : (z == 1) ? Wk : Wv;
    const float*  bias = (z == 0) ? bq : (z == 1) ? bk : bv;
    __bf16*       out  = (z == 0) ? Qo : (z == 1) ? Ko : Vo;
    const bool vtrans = (z == 2);

    const int wave = threadIdx.x >> 5;
    const int lane = threadIdx.x & 31;
    const int half = (lane < 16) ? 0 : 1;
    const int l    = lane & 15;

    const int m0 = blockIdx.y * 64;
    const int n0 = (blockIdx.x * 8 + wave) * 64;

    v8f acc[4][4];
    #pragma unroll
    for (int mi = 0; mi < 4; ++mi)
        #pragma unroll
        for (int ni = 0; ni < 4; ++ni) acc[mi][ni] = (v8f){};

    for (int k0 = 0; k0 < DD; k0 += 32) {
        v16bf af[4], bf[4];
        #pragma unroll
        for (int mi = 0; mi < 4; ++mi)
            af[mi] = load_afrag(X + (size_t)(m0 + mi * 16 + l) * DD + k0 + half * 8);
        #pragma unroll
        for (int ni = 0; ni < 4; ++ni)
            bf[ni] = load_bfrag(W + (size_t)(n0 + ni * 16 + l) * DD + k0 + half * 8);
        #pragma unroll
        for (int mi = 0; mi < 4; ++mi)
            #pragma unroll
            for (int ni = 0; ni < 4; ++ni)
                acc[mi][ni] = wmma_bf16(af[mi], bf[ni], acc[mi][ni]);
    }

    #pragma unroll
    for (int ni = 0; ni < 4; ++ni) {
        const int n  = n0 + ni * 16 + l;
        const float bn = bias[n];
        const int h  = n >> 6;          // n / HD
        const int hd = n & 63;
        #pragma unroll
        for (int mi = 0; mi < 4; ++mi) {
            #pragma unroll
            for (int r = 0; r < 8; ++r) {
                const int m = m0 + mi * 16 + r + half * 8;
                const int b = m >> 11;  // m / S
                const int s = m & 2047;
                const float val = acc[mi][ni][r] + bn;
                const size_t idx = vtrans
                    ? ((size_t)(b * HH + h) * HDD + hd) * SS + s    // [B,H,HD,S]
                    : ((size_t)(b * HH + h) * SS + s) * HDD + hd;   // [B,H,S,HD]
                out[idx] = (__bf16)val;
            }
        }
    }
}

// ---------------------------------------------------------------------------
// Output projection: out[m,n] = sum_k AO[m,k]*Wo[n,k] + bo[n]  (f32 out)
// Same 64x64 register tiling.
// ---------------------------------------------------------------------------
__global__ __launch_bounds__(256) void
gemm_out_f32(const __bf16* __restrict__ A, const __bf16* __restrict__ W,
             const float* __restrict__ bias, float* __restrict__ out) {
    const int wave = threadIdx.x >> 5;
    const int lane = threadIdx.x & 31;
    const int half = (lane < 16) ? 0 : 1;
    const int l    = lane & 15;

    const int m0 = blockIdx.y * 64;
    const int n0 = (blockIdx.x * 8 + wave) * 64;

    v8f acc[4][4];
    #pragma unroll
    for (int mi = 0; mi < 4; ++mi)
        #pragma unroll
        for (int ni = 0; ni < 4; ++ni) acc[mi][ni] = (v8f){};

    for (int k0 = 0; k0 < DD; k0 += 32) {
        v16bf af[4], bf[4];
        #pragma unroll
        for (int mi = 0; mi < 4; ++mi)
            af[mi] = load_afrag(A + (size_t)(m0 + mi * 16 + l) * DD + k0 + half * 8);
        #pragma unroll
        for (int ni = 0; ni < 4; ++ni)
            bf[ni] = load_bfrag(W + (size_t)(n0 + ni * 16 + l) * DD + k0 + half * 8);
        #pragma unroll
        for (int mi = 0; mi < 4; ++mi)
            #pragma unroll
            for (int ni = 0; ni < 4; ++ni)
                acc[mi][ni] = wmma_bf16(af[mi], bf[ni], acc[mi][ni]);
    }

    #pragma unroll
    for (int ni = 0; ni < 4; ++ni) {
        const int n  = n0 + ni * 16 + l;
        const float bn = bias[n];
        #pragma unroll
        for (int mi = 0; mi < 4; ++mi) {
            #pragma unroll
            for (int r = 0; r < 8; ++r) {
                const int m = m0 + mi * 16 + r + half * 8;
                out[(size_t)m * DD + n] = acc[mi][ni][r] + bn;
            }
        }
    }
}

// ---------------------------------------------------------------------------
// Attention: one block (8 waves) per (b, h, 16-query tile).
//   Step 1: scores = Q_tile @ K^T * 1/8  -> LDS (bf16, 16x2048)
//   Step 2: per row: bisection top-k threshold, masked softmax -> P (bf16 LDS)
//   Step 3: O = P @ V^T-layout  (waves 0..3, one 16-wide HD tile each;
//           V stored [B,H,HD,S] so B-fragments are contiguous b128 loads)
// ---------------------------------------------------------------------------
__global__ __launch_bounds__(256) void
attn_kernel(const __bf16* __restrict__ Q, const __bf16* __restrict__ Kmat,
            const __bf16* __restrict__ VT, __bf16* __restrict__ AO) {
    __shared__ __align__(16) __bf16 sc[16][SS];   // 64 KB scores
    __shared__ __align__(16) __bf16 pb[16][SS];   // 64 KB probabilities

    const int blk = blockIdx.x;
    const int qt  = blk & 127;          // S/16 = 128 query tiles
    const int h   = (blk >> 7) & 15;
    const int b   = blk >> 11;
    const int q0  = qt * 16;

    const size_t headBase = (size_t)(b * HH + h) * SS * HDD;
    const __bf16* Qh  = Q    + headBase;
    const __bf16* Kh  = Kmat + headBase;
    const __bf16* VTh = VT   + headBase;   // [HD][S]

    const int wave = threadIdx.x >> 5;
    const int lane = threadIdx.x & 31;
    const int half = (lane < 16) ? 0 : 1;
    const int l    = lane & 15;

    // ---- Step 1: scores -------------------------------------------------
    {
        const __bf16* qrow = Qh + (size_t)(q0 + l) * HDD + half * 8;
        v16bf aq0 = cat16(*(const v8bf*)(qrow +  0), *(const v8bf*)(qrow + 16));
        v16bf aq1 = cat16(*(const v8bf*)(qrow + 32), *(const v8bf*)(qrow + 48));

        for (int t = 0; t < 16; ++t) {
            const int key0 = (wave * 16 + t) * 16;   // 8 waves x 16 tiles x 16 keys
            const __bf16* krow = Kh + (size_t)(key0 + l) * HDD + half * 8;
            v16bf b0 = load_bfrag(krow);
            v16bf b1 = load_bfrag(krow + 32);
            v8f c = {};
            c = wmma_bf16(aq0, b0, c);
            c = wmma_bf16(aq1, b1, c);
            #pragma unroll
            for (int r = 0; r < 8; ++r) {
                int m = r + half * 8;
                sc[m][key0 + l] = (__bf16)(c[r] * 0.125f);   // 1/sqrt(64)
            }
        }
    }
    __syncthreads();

    // ---- Step 2: top-k threshold + softmax (2 rows / wave) -------------
    for (int rr = 0; rr < 2; ++rr) {
        const int row = wave * 2 + rr;
        float x[64];
        #pragma unroll
        for (int j = 0; j < 64; ++j) x[j] = (float)sc[row][lane + j * 32];

        float mx = -3.0e38f, mn = 3.0e38f;
        #pragma unroll
        for (int j = 0; j < 64; ++j) { mx = fmaxf(mx, x[j]); mn = fminf(mn, x[j]); }
        mx = wred_max(mx);
        mn = wred_min(mn);

        // bisection for the KK-th largest value
        float lo = mn, hi = mx;
        for (int it = 0; it < 30; ++it) {
            float mid = 0.5f * (lo + hi);
            int cnt = 0;
            #pragma unroll
            for (int j = 0; j < 64; ++j) cnt += (x[j] >= mid) ? 1 : 0;
            cnt = wred_sum_i(cnt);
            if (cnt >= KK) lo = mid; else hi = mid;
        }
        const float thr = lo;

        float sum = 0.0f;
        #pragma unroll
        for (int j = 0; j < 64; ++j) {
            float v = x[j] - mx + ((x[j] < thr) ? -10000.0f : 0.0f);
            sum += __expf(v);
        }
        sum = wred_sum(sum);
        const float inv = 1.0f / sum;

        #pragma unroll
        for (int j = 0; j < 64; ++j) {
            float v = x[j] - mx + ((x[j] < thr) ? -10000.0f : 0.0f);
            pb[row][lane + j * 32] = (__bf16)(__expf(v) * inv);
        }
    }
    __syncthreads();

    // ---- Step 3: O = P @ V (waves 0..3) --------------------------------
    if (wave < 4) {
        const int hd0 = wave * 16;
        // B-fragment row pointer: lane = hd column, contiguous keys
        const __bf16* vrow = VTh + (size_t)(hd0 + l) * SS;
        v8f c = {};
        for (int kt = 0; kt < 64; ++kt) {
            const int key0 = kt * 32;
            const __bf16* prow = &pb[l][key0 + half * 8];
            v16bf a = cat16(*(const v8bf*)(prow), *(const v8bf*)(prow + 16));
            v16bf bb = load_bfrag(vrow + key0 + half * 8);
            c = wmma_bf16(a, bb, c);
        }
        #pragma unroll
        for (int r = 0; r < 8; ++r) {
            int m = r + half * 8;
            int s = q0 + m;
            // [B,S,D] with D = H*HD
            AO[((size_t)(b * SS + s) * DD) + h * HDD + hd0 + l] = (__bf16)c[r];
        }
    }
}

// ---------------------------------------------------------------------------
// Host launcher
// ---------------------------------------------------------------------------
extern "C" void kernel_launch(void* const* d_in, const int* in_sizes, int n_in,
                              void* d_out, int out_size, void* d_ws, size_t ws_size,
                              hipStream_t stream) {
    const float* X  = (const float*)d_in[0];
    const float* Wq = (const float*)d_in[1];
    const float* bq = (const float*)d_in[2];
    const float* Wk = (const float*)d_in[3];
    const float* bk = (const float*)d_in[4];
    const float* Wv = (const float*)d_in[5];
    const float* bv = (const float*)d_in[6];
    const float* Wo = (const float*)d_in[7];
    const float* bo = (const float*)d_in[8];
    float* out = (float*)d_out;

    const size_t nX = (size_t)BB * SS * DD;   // 4,194,304
    const size_t nW = (size_t)DD * DD;        // 1,048,576

    char* p = (char*)d_ws;
    auto take = [&](size_t bytes) {
        char* r = p;
        p += (bytes + 255) & ~(size_t)255;
        return r;
    };
    __bf16* Xb  = (__bf16*)take(nX * 2);
    __bf16* Wqb = (__bf16*)take(nW * 2);
    __bf16* Wkb = (__bf16*)take(nW * 2);
    __bf16* Wvb = (__bf16*)take(nW * 2);
    __bf16* Wob = (__bf16*)take(nW * 2);
    __bf16* Qb  = (__bf16*)take(nX * 2);      // [B,H,S,HD]
    __bf16* Kb  = (__bf16*)take(nX * 2);      // [B,H,S,HD]
    __bf16* Vb  = (__bf16*)take(nX * 2);      // [B,H,HD,S] (transposed)
    __bf16* AOb = (__bf16*)take(nX * 2);      // [B,S,D]

    // 1) convert everything to bf16
    cvt_f32_bf16<<<dim3((unsigned)((nX + 255) / 256)), dim3(256), 0, stream>>>(X,  Xb,  (int)nX);
    cvt_f32_bf16<<<dim3((unsigned)((nW + 255) / 256)), dim3(256), 0, stream>>>(Wq, Wqb, (int)nW);
    cvt_f32_bf16<<<dim3((unsigned)((nW + 255) / 256)), dim3(256), 0, stream>>>(Wk, Wkb, (int)nW);
    cvt_f32_bf16<<<dim3((unsigned)((nW + 255) / 256)), dim3(256), 0, stream>>>(Wv, Wvb, (int)nW);
    cvt_f32_bf16<<<dim3((unsigned)((nW + 255) / 256)), dim3(256), 0, stream>>>(Wo, Wob, (int)nW);

    // 2) fused Q/K/V projections (x @ W.T + b)
    //    grid: x = N/(8 waves*64) = 2, y = M/64 = 64, z = 3 matrices
    dim3 qkvGrid(DD / 512, (BB * SS) / 64, 3);
    gemm_qkv_fused<<<qkvGrid, dim3(256), 0, stream>>>(Xb, Wqb, Wkb, Wvb,
                                                      bq, bk, bv, Qb, Kb, Vb);

    // 3) sparse attention
    attn_kernel<<<dim3(BB * HH * (SS / 16)), dim3(256), 0, stream>>>(Qb, Kb, Vb, AOb);

    // 4) output projection -> f32
    dim3 outGrid(DD / 512, (BB * SS) / 64);
    gemm_out_f32<<<outGrid, dim3(256), 0, stream>>>(AOb, Wob, bo, out);
}
